// LSTM_Model_76845554860373
// MI455X (gfx1250) — matrix operand order repaired and checked by reading, not compiled, not run on hardware
//
#include <hip/hip_runtime.h>
#include <hip/hip_bf16.h>

#define SEQ   4096
#define VOCAB 32000
#define EMB   64
#define HID   16
#define VTILES (VOCAB / 16)   // 2000

typedef __attribute__((ext_vector_type(16))) _Float16 v16h;
typedef __attribute__((ext_vector_type(8)))  _Float16 v8h;
typedef __attribute__((ext_vector_type(8)))  float    v8f;

// ---------------------------------------------------------------------------
// Kernel 1: W_lin f32 -> f16 (so the GEMM passes stream half the bytes)
// ---------------------------------------------------------------------------
__global__ __launch_bounds__(256) void cvt_wlin_kernel(
    const float* __restrict__ W_lin, _Float16* __restrict__ wlin16) {
  int i = blockIdx.x * 256 + threadIdx.x;     // exactly VOCAB*HID = 512000
  wlin16[i] = (_Float16)W_lin[i];
}

// ---------------------------------------------------------------------------
// Kernel 2: xg[t][g] = emb[lookups[t]] . W_ih[g] + b_ih[g] + b_hh[g]
// ---------------------------------------------------------------------------
__global__ __launch_bounds__(256) void xg_kernel(
    const int*   __restrict__ lookups,
    const float* __restrict__ emb,
    const float* __restrict__ W_ih,
    const float* __restrict__ b_ih,
    const float* __restrict__ b_hh,
    float*       __restrict__ xg) {
  int idx = blockIdx.x * 256 + threadIdx.x;   // exactly SEQ*64
  int t = idx >> 6, g = idx & 63;
  int row = lookups[t];
  const float4* e4 = (const float4*)(emb + (long)row * EMB);
  const float4* w4 = (const float4*)(W_ih + g * EMB);
  float acc = b_ih[g] + b_hh[g];
#pragma unroll
  for (int q = 0; q < EMB / 4; ++q) {
    float4 e = e4[q], w = w4[q];
    acc = fmaf(e.x, w.x, acc); acc = fmaf(e.y, w.y, acc);
    acc = fmaf(e.z, w.z, acc); acc = fmaf(e.w, w.w, acc);
  }
  xg[idx] = acc;
}

// ---------------------------------------------------------------------------
// Kernel 3: sequential LSTM scan. One wave32. Lane L owns gates L and L+32.
// Gate order (PyTorch): i=0..15, f=16..31, g=32..47, o=48..63.
// ---------------------------------------------------------------------------
__device__ __forceinline__ float sigmoid_f(float x) {
  return 1.0f / (1.0f + __expf(-x));
}
__device__ __forceinline__ float tanh_f(float x) {
  return 2.0f / (1.0f + __expf(-2.0f * x)) - 1.0f;
}

__global__ __launch_bounds__(32) void lstm_scan_kernel(
    const float* __restrict__ xg,      // [SEQ][64]
    const float* __restrict__ W_hh,    // [64][16]
    _Float16*    __restrict__ hs16) {  // [SEQ][16]
  const int L = threadIdx.x;           // 0..31
  float w0[HID], w1[HID];
#pragma unroll
  for (int j = 0; j < HID; ++j) {
    w0[j] = W_hh[L * HID + j];          // row L        (i/f gates)
    w1[j] = W_hh[(L + 32) * HID + j];   // row L+32     (g/o gates)
  }

  __shared__ float hsh[HID];
  if (L < HID) hsh[L] = 0.0f;
  float c = 0.0f;                      // lanes 0..15 only
  __syncthreads();

  for (int t = 0; t < SEQ; ++t) {
    float a0 = xg[t * 64 + L];
    float a1 = xg[t * 64 + L + 32];
    float h[HID];
#pragma unroll
    for (int j = 0; j < HID; ++j) h[j] = hsh[j];
    float acc0 = a0, acc1 = a1;
#pragma unroll
    for (int j = 0; j < HID; ++j) {
      acc0 = fmaf(w0[j], h[j], acc0);
      acc1 = fmaf(w1[j], h[j], acc1);
    }
    float act0 = sigmoid_f(acc0);                       // i (lanes<16) / f
    float act1 = (L < HID) ? tanh_f(acc1)               // g
                           : sigmoid_f(acc1);           // o
    int src = (L & 15) + 16;
    float fg = __shfl(act0, src, 32);                   // f[j] for lanes<16
    float og = __shfl(act1, src, 32);                   // o[j] for lanes<16
    __syncthreads();                                    // WAR on hsh
    if (L < HID) {
      c = fmaf(fg, c, act0 * act1);                     // c = f*c + i*g
      float hn = og * tanh_f(c);
      hsh[L] = hn;
      hs16[t * HID + L] = (_Float16)hn;
    }
    __syncthreads();                                    // RAW on hsh
  }
}

// ---------------------------------------------------------------------------
// Kernel 4: logits + log_softmax via v_wmma_f32_16x16x32_f16.
// Block = 256 threads (8 waves), owns 16 timesteps. Two passes over vocab:
//   pass 1: online (max, sum-exp) per row; pass 2: recompute, write x - logZ.
// ---------------------------------------------------------------------------
__global__ __launch_bounds__(256) void logits_lsm_kernel(
    const _Float16* __restrict__ hs16,    // [SEQ][16]
    const _Float16* __restrict__ wlin16,  // [VOCAB][16]
    const float*    __restrict__ b_lin,   // [VOCAB]
    float*          __restrict__ out) {   // [SEQ][VOCAB]
  const int tid  = threadIdx.x;
  const int lane = tid & 31;
  const int wave = tid >> 5;              // 0..7
  const int t0   = blockIdx.x * 16;
  const int col  = lane & 15;             // output column within tile
  const int hi   = lane >> 4;             // 0: rows 0-7, 1: rows 8-15

  __shared__ float lm[16][128];
  __shared__ float ls[16][128];
  __shared__ float lz[16];

  // A operand (constant over vocab): lane<16 -> hs[t0+col][0..15] (K=0..15),
  // lane>=16 -> zero pad (K=16..31).
  v16h a = {};
  {
    if (!hi) a = *(const v16h*)(hs16 + (t0 + col) * HID);
  }

  float m[8], s[8];
#pragma unroll
  for (int r = 0; r < 8; ++r) { m[r] = -1e30f; s[r] = 0.0f; }

  // ---- pass 1: online logsumexp, nothing stored ----
  for (int vt = wave; vt < VTILES; vt += 8) {
    int vb = vt * 16;
    if (vt + 8 < VTILES)
      __builtin_prefetch(wlin16 + (vt + 8) * 16 * HID + col * HID, 0, 0);
    v16h b = *(const v16h*)(wlin16 + (vb + col) * HID);  // W_lin row vb+col
    if (hi) { v16h zb = {}; b = zb; }                    // K=16..31 pad
    float bl = b_lin[vb + col];
    v8f cc;
#pragma unroll
    for (int r = 0; r < 8; ++r) cc[r] = bl;
    v8f d = __builtin_amdgcn_wmma_f32_16x16x32_f16(
        false, a, false, b, (short)0, cc, false, false);
#pragma unroll
    for (int r = 0; r < 8; ++r) {
      float x  = d[r];
      float mn = fmaxf(m[r], x);
      s[r] = s[r] * __expf(m[r] - mn) + __expf(x - mn);
      m[r] = mn;
    }
  }

  // merge partials: row R gets 8 waves * 16 lanes = 128 (m,s) pairs
#pragma unroll
  for (int r = 0; r < 8; ++r) {
    lm[hi * 8 + r][wave * 16 + col] = m[r];
    ls[hi * 8 + r][wave * 16 + col] = s[r];
  }
  __syncthreads();
  if (tid < 16) {
    float M = -1e30f;
    for (int p = 0; p < 128; ++p) M = fmaxf(M, lm[tid][p]);
    float S = 0.0f;
    for (int p = 0; p < 128; ++p) S += ls[tid][p] * __expf(lm[tid][p] - M);
    lz[tid] = M + __logf(S);                // logZ = max + log(sum exp)
  }
  __syncthreads();
  float lzr[8];
#pragma unroll
  for (int r = 0; r < 8; ++r) lzr[r] = lz[hi * 8 + r];

  // ---- pass 2: recompute tiles, write log_softmax ----
  for (int vt = wave; vt < VTILES; vt += 8) {
    int vb = vt * 16;
    v16h b = *(const v16h*)(wlin16 + (vb + col) * HID);
    if (hi) { v16h zb = {}; b = zb; }
    float bl = b_lin[vb + col];
    v8f cc;
#pragma unroll
    for (int r = 0; r < 8; ++r) cc[r] = bl;
    v8f d = __builtin_amdgcn_wmma_f32_16x16x32_f16(
        false, a, false, b, (short)0, cc, false, false);
    int vcol = vb + col;
#pragma unroll
    for (int r = 0; r < 8; ++r) {
      long row = t0 + hi * 8 + r;
      out[row * (long)VOCAB + vcol] = d[r] - lzr[r];
    }
  }
}

// ---------------------------------------------------------------------------
extern "C" void kernel_launch(void* const* d_in, const int* in_sizes, int n_in,
                              void* d_out, int out_size, void* d_ws, size_t ws_size,
                              hipStream_t stream) {
  const int*   lookups = (const int*)  d_in[0];
  const float* emb     = (const float*)d_in[1];
  const float* W_ih    = (const float*)d_in[2];
  const float* W_hh    = (const float*)d_in[3];
  const float* b_ih    = (const float*)d_in[4];
  const float* b_hh    = (const float*)d_in[5];
  const float* W_lin   = (const float*)d_in[6];
  const float* b_lin   = (const float*)d_in[7];
  float* out = (float*)d_out;

  char* ws = (char*)d_ws;
  float*    xg     = (float*)ws;                                   // 1,048,576 B
  _Float16* hs16   = (_Float16*)(ws + 1048576);                    //   131,072 B
  _Float16* wlin16 = (_Float16*)(ws + 1048576 + 131072);           // 1,024,000 B

  cvt_wlin_kernel<<<dim3(VOCAB * HID / 256), dim3(256), 0, stream>>>(W_lin, wlin16);
  xg_kernel<<<dim3(SEQ * 64 / 256), dim3(256), 0, stream>>>(
      lookups, emb, W_ih, b_ih, b_hh, xg);
  lstm_scan_kernel<<<dim3(1), dim3(32), 0, stream>>>(xg, W_hh, hs16);
  logits_lsm_kernel<<<dim3(SEQ / 16), dim3(256), 0, stream>>>(
      hs16, wlin16, b_lin, out);
}
